// FasterMultiHeadAttention_37795712205107
// MI455X (gfx1250) — matrix operand order repaired and checked by reading, not compiled, not run on hardware
//
#include <hip/hip_runtime.h>

typedef __bf16 bf16_t;
typedef __attribute__((ext_vector_type(16))) __bf16 v16bf;
typedef __attribute__((ext_vector_type(8)))  __bf16 v8bf;
typedef __attribute__((ext_vector_type(8)))  float  v8f;
typedef __attribute__((ext_vector_type(4)))  unsigned int u32x4;
typedef __attribute__((ext_vector_type(8)))  unsigned int u32x8;

#define BB 2
#define SS 2048
#define DD 1024
#define HH 16
#define HD 64
#define MTOT (BB * SS)          // 4096 rows
#define N_QKV (3 * DD)          // 3072

// ---------------------------------------------------------------------------
// WMMA fragment loaders (ISA 7.12.2 VGPR layouts, wave32)
// ---------------------------------------------------------------------------

// A matrix 16x32 (MxK) bf16, row-major source with leading dim lda.
// lane L: row m = L&15; K base = k0 + (L>=16 ? 8 : 0);
// elements 0..7 = K[base..base+7], elements 8..15 = K[base+16..base+23].
__device__ __forceinline__ v16bf load_a_frag(const bf16_t* A, int lda,
                                             int row0, int k0, int lane) {
  const int half = lane >> 4;
  const int m    = lane & 15;
  const bf16_t* base = A + (size_t)(row0 + m) * lda + k0 + half * 8;
  union { v16bf v; v8bf h[2]; } u;
  u.h[0] = *(const v8bf*)(base);
  u.h[1] = *(const v8bf*)(base + 16);
  return u.v;
}

// B matrix 32x16 (KxN) bf16 from a TRANSPOSED source Bt[N][K] (row-major, ld=ldb).
// lane L: col n = L&15; K base = k0 + (L>=16 ? 8 : 0);
// elements 0..7 = K[base..base+7], elements 8..15 = K[base+16..base+23].
__device__ __forceinline__ v16bf load_bt_frag(const bf16_t* __restrict__ Bt, int ldb,
                                              int n0, int k0, int lane) {
  const int half = lane >> 4;
  const int n    = lane & 15;
  const bf16_t* base = Bt + (size_t)(n0 + n) * ldb + k0 + half * 8;
  union { v16bf v; v8bf h[2]; } u;
  u.h[0] = *(const v8bf*)(base);
  u.h[1] = *(const v8bf*)(base + 16);
  return u.v;
}

__device__ __forceinline__ v8f wmma_bf16(v16bf a, v16bf b, v8f c) {
  return __builtin_amdgcn_wmma_f32_16x16x32_bf16(false, a, false, b, (short)0, c,
                                                 false, false);
}

// ---------------------------------------------------------------------------
// Tensor Data Mover: 2D bf16 tile load Global -> LDS (CDNA5 ISA).
// Group0: count=1 | lds_addr | 57b global_addr | type=2.
// Group1: data_size=1 (2B), tensor_dim0/1, tile_dim0/1, tensor_dim0_stride.
// Tile lands packed row-major (tile_d0 elements per row) at lds_byte_addr.
// ---------------------------------------------------------------------------
__device__ __forceinline__ void tdm_load_tile_bf16(unsigned lds_byte_addr,
                                                   const bf16_t* gsrc,
                                                   unsigned tensor_d0, unsigned tensor_d1,
                                                   unsigned stride0,
                                                   unsigned tile_d0, unsigned tile_d1) {
  const unsigned long long ga = (unsigned long long)(size_t)gsrc;
  u32x4 g0;
  g0[0] = 1u;                                              // count=1, user D#
  g0[1] = lds_byte_addr;                                   // lds_addr[31:0]
  g0[2] = (unsigned)(ga & 0xffffffffu);                    // global_addr[31:0]
  g0[3] = (unsigned)((ga >> 32) & 0x1ffffffu) | (2u << 30);// addr[56:32] | type=2
  u32x8 g1;
  g1[0] = (1u << 16);                                      // wg_mask=0, data_size=2B
  g1[1] = (tensor_d0 & 0xffffu) << 16;                     // abar=0 | dim0[15:0]
  g1[2] = (tensor_d0 >> 16) | ((tensor_d1 & 0xffffu) << 16);
  g1[3] = (tensor_d1 >> 16) | (tile_d0 << 16);             // dim1[31:16] | tile_d0
  g1[4] = tile_d1;                                         // tile_d1 | tile_d2=0
  g1[5] = stride0;                                         // dim0_stride[31:0]
  g1[6] = 0u;                                              // stride0 hi | stride1 lo
  g1[7] = 0u;
  asm volatile("tensor_load_to_lds %0, %1" :: "s"(g0), "s"(g1) : "memory");
}

__device__ __forceinline__ unsigned lds_off32(const void* p) {
  return (unsigned)(size_t)p;   // LDS aperture: addr[31:0] is the LDS byte offset
}

// ---------------------------------------------------------------------------
// Prep: f32 -> bf16 cast, and tiled transpose W[K,N] -> Wt[N,K] bf16
// ---------------------------------------------------------------------------
__global__ void __launch_bounds__(256, 1)
k_cast_bf16(const float* __restrict__ src, bf16_t* __restrict__ dst, int n) {
  int i = blockIdx.x * blockDim.x + threadIdx.x;
  if (i < n) dst[i] = (bf16_t)src[i];
}

__global__ void __launch_bounds__(256, 1)
k_transpose_bf16(const float* __restrict__ W, bf16_t* __restrict__ Wt, int K, int N) {
  __shared__ float tile[32][33];
  const int k0 = blockIdx.x * 32;
  const int n0 = blockIdx.y * 32;
  const int tx = threadIdx.x & 31;
  const int ty = threadIdx.x >> 5;          // 0..7 (256 threads)
  for (int i = ty; i < 32; i += 8)
    tile[i][tx] = W[(size_t)(k0 + i) * N + (n0 + tx)];
  __syncthreads();
  for (int i = ty; i < 32; i += 8)
    Wt[(size_t)(n0 + i) * K + (k0 + tx)] = (bf16_t)tile[tx][i];
}

// ---------------------------------------------------------------------------
// QKV projection: xb[M,D] @ Wt[3D,D]^T.
// Block = 128 threads (4 waves), block tile 32(M) x 256(N); wave tile 32x64.
// A (32x32 bf16) staged via TDM double-buffered in LDS; B direct from global.
// Epilogue scatters into Q/K (head-major, Q pre-scaled 1/8) and Vt [B,H,HD,S].
// grid: (M/32, 3D/256)
// ---------------------------------------------------------------------------
__global__ void __launch_bounds__(128, 1)
k_qkv_gemm(const bf16_t* __restrict__ xb, const bf16_t* __restrict__ Wt,
           const float* __restrict__ bias,
           bf16_t* __restrict__ Q, bf16_t* __restrict__ Kh,
           bf16_t* __restrict__ Vt) {
  __shared__ __align__(16) bf16_t atile[2][32 * 32];
  const int lane = threadIdx.x & 31;
  const int wave = threadIdx.x >> 5;
  const int m0   = blockIdx.x * 32;
  const int n0   = blockIdx.y * 256 + wave * 64;
  const int half = lane >> 4;
  const int KSTEPS = DD / 32;

  v8f acc[2][4];
  #pragma unroll
  for (int ms = 0; ms < 2; ++ms)
    #pragma unroll
    for (int t = 0; t < 4; ++t) acc[ms][t] = v8f{0};

  if (wave == 0)
    tdm_load_tile_bf16(lds_off32(&atile[0][0]), xb + (size_t)m0 * DD,
                       DD, MTOT, DD, 32, 32);

  for (int ks = 0; ks < KSTEPS; ++ks) {
    __builtin_amdgcn_s_wait_tensorcnt(0);              // scalar; free for waves 1-3
    __syncthreads();                                   // tile ks visible; buf free
    if (wave == 0 && ks + 1 < KSTEPS)
      tdm_load_tile_bf16(lds_off32(&atile[(ks + 1) & 1][0]),
                         xb + (size_t)m0 * DD + (ks + 1) * 32,
                         DD, MTOT, DD, 32, 32);
    const bf16_t* at = &atile[ks & 1][0];
    v16bf a0 = load_a_frag(at, 32, 0,  0, lane);
    v16bf a1 = load_a_frag(at, 32, 16, 0, lane);
    __builtin_prefetch(Wt + (size_t)(n0 + (lane & 15)) * DD + ks * 32 + 128, 0, 1);
    #pragma unroll
    for (int t = 0; t < 4; ++t) {
      v16bf b = load_bt_frag(Wt, DD, n0 + t * 16, ks * 32, lane);
      acc[0][t] = wmma_bf16(a0, b, acc[0][t]);
      acc[1][t] = wmma_bf16(a1, b, acc[1][t]);
    }
  }

  #pragma unroll
  for (int t = 0; t < 4; ++t) {
    const int c  = n0 + t * 16 + (lane & 15);
    const float bv = bias[c];
    #pragma unroll
    for (int ms = 0; ms < 2; ++ms) {
      #pragma unroll
      for (int r = 0; r < 8; ++r) {
        const int mg = m0 + ms * 16 + r + half * 8;
        const int b  = mg / SS;
        const int s  = mg - b * SS;
        const float v = acc[ms][t][r] + bv;
        if (c < DD) {                       // Q (scaled by 1/sqrt(HD))
          const int h = c >> 6, hd = c & 63;
          Q[(((size_t)b * HH + h) * SS + s) * HD + hd] = (bf16_t)(v * 0.125f);
        } else if (c < 2 * DD) {            // K
          const int c2 = c - DD, h = c2 >> 6, hd = c2 & 63;
          Kh[(((size_t)b * HH + h) * SS + s) * HD + hd] = (bf16_t)v;
        } else {                            // V transposed: [B,H,HD,S]
          const int c2 = c - 2 * DD, h = c2 >> 6, hd = c2 & 63;
          Vt[(((size_t)b * HH + h) * HD + hd) * SS + s] = (bf16_t)v;
        }
      }
    }
  }
}

// ---------------------------------------------------------------------------
// Flash attention: one wave per 16-query tile, streaming 32 keys per step.
// grid: (B*H, S/64), block 128 (4 waves)
// ---------------------------------------------------------------------------
__global__ void __launch_bounds__(128, 1)
k_attn(const bf16_t* __restrict__ Q, const bf16_t* __restrict__ Kh,
       const bf16_t* __restrict__ Vt, bf16_t* __restrict__ attnb) {
  __shared__ __align__(16) bf16_t pbuf[4][16 * 32];
  const int lane = threadIdx.x & 31;
  const int wave = threadIdx.x >> 5;
  const int half = lane >> 4;
  const int bh = blockIdx.x;
  const int b  = bh / HH;
  const int h  = bh - b * HH;
  const int q0 = (blockIdx.y * 4 + wave) * 16;

  const bf16_t* Qp  = Q  + (size_t)bh * SS * HD;
  const bf16_t* Kp  = Kh + (size_t)bh * SS * HD;
  const bf16_t* Vtp = Vt + (size_t)bh * HD * SS;
  bf16_t* pb = &pbuf[wave][0];

  v8f acc[4] = {v8f{0}, v8f{0}, v8f{0}, v8f{0}};
  float m8[8], l8[8];
  #pragma unroll
  for (int r = 0; r < 8; ++r) { m8[r] = -1e30f; l8[r] = 0.0f; }

  for (int kc = 0; kc < SS; kc += 32) {
    // ---- scores: Q(16xHD) @ K^T -> two 16x16 tiles (keys kc..+15, kc+16..+31)
    v8f cs0 = v8f{0}, cs1 = v8f{0};
    #pragma unroll
    for (int kk = 0; kk < HD; kk += 32) {
      v16bf aq = load_a_frag(Qp, HD, q0, kk, lane);
      cs0 = wmma_bf16(aq, load_bt_frag(Kp, HD, kc,      kk, lane), cs0);
      cs1 = wmma_bf16(aq, load_bt_frag(Kp, HD, kc + 16, kk, lane), cs1);
    }

    // ---- online softmax (rows at vgpr r + 8*laneHalf; cols across 16 lanes)
    float tmax[8], tsum[8], corr[8], p0[8], p1[8];
    #pragma unroll
    for (int r = 0; r < 8; ++r) tmax[r] = fmaxf(cs0[r], cs1[r]);
    #pragma unroll
    for (int off = 1; off < 16; off <<= 1)
      #pragma unroll
      for (int r = 0; r < 8; ++r) tmax[r] = fmaxf(tmax[r], __shfl_xor(tmax[r], off, 32));
    #pragma unroll
    for (int r = 0; r < 8; ++r) {
      const float mn = fmaxf(m8[r], tmax[r]);
      corr[r] = __expf(m8[r] - mn);
      m8[r]   = mn;
      p0[r]   = __expf(cs0[r] - mn);
      p1[r]   = __expf(cs1[r] - mn);
      tsum[r] = p0[r] + p1[r];
    }
    #pragma unroll
    for (int off = 1; off < 16; off <<= 1)
      #pragma unroll
      for (int r = 0; r < 8; ++r) tsum[r] += __shfl_xor(tsum[r], off, 32);
    #pragma unroll
    for (int r = 0; r < 8; ++r) l8[r] = l8[r] * corr[r] + tsum[r];
    #pragma unroll
    for (int t = 0; t < 4; ++t)
      #pragma unroll
      for (int r = 0; r < 8; ++r) acc[t][r] *= corr[r];

    // ---- stage P (16x32) through LDS: C-layout -> A-fragment reshape
    {
      const int col = lane & 15;
      #pragma unroll
      for (int r = 0; r < 8; ++r) {
        const int row = r + half * 8;
        pb[row * 32 + col]      = (bf16_t)p0[r];
        pb[row * 32 + 16 + col] = (bf16_t)p1[r];
      }
    }
    v16bf ap = load_a_frag(pb, 32, 0, 0, lane);   // same-wave LDS: in-order

    // ---- P(16x32) @ V(32xHD): Vt rows = hd, cols = s (contiguous K)
    #pragma unroll
    for (int t = 0; t < 4; ++t)
      acc[t] = wmma_bf16(ap, load_bt_frag(Vtp, SS, t * 16, kc, lane), acc[t]);
  }

  // ---- normalize and scatter to attnb[B,S,D] (bf16)
  #pragma unroll
  for (int r = 0; r < 8; ++r) l8[r] = 1.0f / l8[r];
  #pragma unroll
  for (int t = 0; t < 4; ++t) {
    const int hd = t * 16 + (lane & 15);
    #pragma unroll
    for (int r = 0; r < 8; ++r) {
      const int row = r + half * 8;
      attnb[((size_t)b * SS + q0 + row) * DD + h * 64 + hd] = (bf16_t)(acc[t][r] * l8[r]);
    }
  }
}

// ---------------------------------------------------------------------------
// Output projection: attnb[M,D] @ Wot[D,D]^T + b_out -> out f32.
// Same TDM-staged 32x256 block tiling as QKV. grid: (M/32, D/256)
// ---------------------------------------------------------------------------
__global__ void __launch_bounds__(128, 1)
k_out_gemm(const bf16_t* __restrict__ attnb, const bf16_t* __restrict__ Wot,
           const float* __restrict__ bias, float* __restrict__ out) {
  __shared__ __align__(16) bf16_t atile[2][32 * 32];
  const int lane = threadIdx.x & 31;
  const int wave = threadIdx.x >> 5;
  const int m0   = blockIdx.x * 32;
  const int n0   = blockIdx.y * 256 + wave * 64;
  const int half = lane >> 4;
  const int KSTEPS = DD / 32;

  v8f acc[2][4];
  #pragma unroll
  for (int ms = 0; ms < 2; ++ms)
    #pragma unroll
    for (int t = 0; t < 4; ++t) acc[ms][t] = v8f{0};

  if (wave == 0)
    tdm_load_tile_bf16(lds_off32(&atile[0][0]), attnb + (size_t)m0 * DD,
                       DD, MTOT, DD, 32, 32);

  for (int ks = 0; ks < KSTEPS; ++ks) {
    __builtin_amdgcn_s_wait_tensorcnt(0);
    __syncthreads();
    if (wave == 0 && ks + 1 < KSTEPS)
      tdm_load_tile_bf16(lds_off32(&atile[(ks + 1) & 1][0]),
                         attnb + (size_t)m0 * DD + (ks + 1) * 32,
                         DD, MTOT, DD, 32, 32);
    const bf16_t* at = &atile[ks & 1][0];
    v16bf a0 = load_a_frag(at, 32, 0,  0, lane);
    v16bf a1 = load_a_frag(at, 32, 16, 0, lane);
    __builtin_prefetch(Wot + (size_t)(n0 + (lane & 15)) * DD + ks * 32 + 128, 0, 1);
    #pragma unroll
    for (int t = 0; t < 4; ++t) {
      v16bf b = load_bt_frag(Wot, DD, n0 + t * 16, ks * 32, lane);
      acc[0][t] = wmma_bf16(a0, b, acc[0][t]);
      acc[1][t] = wmma_bf16(a1, b, acc[1][t]);
    }
  }

  #pragma unroll
  for (int t = 0; t < 4; ++t) {
    const int c  = n0 + t * 16 + (lane & 15);
    const float bv = bias[c];
    #pragma unroll
    for (int ms = 0; ms < 2; ++ms)
      #pragma unroll
      for (int r = 0; r < 8; ++r) {
        const int mg = m0 + ms * 16 + r + half * 8;
        out[(size_t)mg * DD + c] = acc[ms][t][r] + bv;
      }
  }
}

// ---------------------------------------------------------------------------
extern "C" void kernel_launch(void* const* d_in, const int* in_sizes, int n_in,
                              void* d_out, int out_size, void* d_ws, size_t ws_size,
                              hipStream_t stream) {
  const float* x     = (const float*)d_in[0];
  const float* W_qkv = (const float*)d_in[1];
  const float* b_qkv = (const float*)d_in[2];
  const float* W_out = (const float*)d_in[3];
  const float* b_out = (const float*)d_in[4];
  float* out = (float*)d_out;

  char* ws = (char*)d_ws;
  size_t off = 0;
  bf16_t* xb    = (bf16_t*)(ws + off); off += (size_t)MTOT * DD * sizeof(bf16_t);      // 8 MB
  bf16_t* Wqt   = (bf16_t*)(ws + off); off += (size_t)N_QKV * DD * sizeof(bf16_t);     // 6 MB
  bf16_t* Wot   = (bf16_t*)(ws + off); off += (size_t)DD * DD * sizeof(bf16_t);        // 2 MB
  bf16_t* Q     = (bf16_t*)(ws + off); off += (size_t)BB * HH * SS * HD * sizeof(bf16_t);
  bf16_t* Kh    = (bf16_t*)(ws + off); off += (size_t)BB * HH * SS * HD * sizeof(bf16_t);
  bf16_t* Vt    = (bf16_t*)(ws + off); off += (size_t)BB * HH * HD * SS * sizeof(bf16_t);
  bf16_t* attnb = (bf16_t*)(ws + off); off += (size_t)MTOT * DD * sizeof(bf16_t);      // 48 MB

  // 1) prep
  {
    const int n = MTOT * DD;
    k_cast_bf16<<<(n + 255) / 256, 256, 0, stream>>>(x, xb, n);
  }
  k_transpose_bf16<<<dim3(DD / 32, N_QKV / 32), 256, 0, stream>>>(W_qkv, Wqt, DD, N_QKV);
  k_transpose_bf16<<<dim3(DD / 32, DD / 32),   256, 0, stream>>>(W_out, Wot, DD, DD);

  // 2) QKV projection (TDM-staged A, scale Q, head-major scatter, V transposed)
  k_qkv_gemm<<<dim3(MTOT / 32, N_QKV / 256), 128, 0, stream>>>(xb, Wqt, b_qkv, Q, Kh, Vt);

  // 3) flash attention
  k_attn<<<dim3(BB * HH, SS / 64), 128, 0, stream>>>(Q, Kh, Vt, attnb);

  // 4) output projection (TDM-staged A)
  k_out_gemm<<<dim3(MTOT / 32, DD / 256), 128, 0, stream>>>(attnb, Wot, b_out, out);
}
